// RnnDecoder_13048110646094
// MI455X (gfx1250) — matrix operand order repaired and checked by reading, not compiled, not run on hardware
//
#include <hip/hip_runtime.h>

// GRU decoder: B=64, T=512, H=1024.
// Phase 1: gi[t,b,3H] = m_enc @ W_ih^T + b_ih  (bf16 WMMA, f32 accumulate)
// Phase 2: 512 fused step kernels: gh = h @ W_hh^T (3 gate tiles per wave),
//          gates + state update in-register, double-buffered h.

#define B_  64
#define T_  512
#define H_  1024
#define G_  3072   // 3*H
#define K_  1024

typedef __bf16 v16bf __attribute__((ext_vector_type(16)));
typedef __bf16 v8bf  __attribute__((ext_vector_type(8)));
typedef float  v8f   __attribute__((ext_vector_type(8)));

static __device__ __forceinline__ unsigned short f2bf(float f) {
  unsigned u = __float_as_uint(f);
  u += 0x7FFFu + ((u >> 16) & 1u);      // round-to-nearest-even
  return (unsigned short)(u >> 16);
}
static __device__ __forceinline__ float bf2f(unsigned short h) {
  return __uint_as_float(((unsigned)h) << 16);
}
static __device__ __forceinline__ float sig_(float x) {
  return 1.0f / (1.0f + __expf(-x));
}
static __device__ __forceinline__ float tanh_(float x) {
  return 2.0f / (1.0f + __expf(-2.0f * x)) - 1.0f;
}

// Load one 16x32 bf16 A-tile fragment for this lane (contiguous k order, same as B):
// row = m_base + lane%16; K runs [kb + half*16, +16).
static __device__ __forceinline__ v16bf load_a_frag(const unsigned short* __restrict__ rowp,
                                                    int kb, int half) {
  v8bf a0 = *(const v8bf*)(rowp + kb + half * 16);
  v8bf a1 = *(const v8bf*)(rowp + kb + half * 16 + 8);
  return __builtin_shufflevector(a0, a1, 0, 1, 2, 3, 4, 5, 6, 7,
                                         8, 9, 10, 11, 12, 13, 14, 15);
}

// ---------------------------------------------------------------------------
// fp32 -> bf16 conversion (4 elems/thread)
__global__ __launch_bounds__(256) void cvt_f32_bf16_kernel(
    const float* __restrict__ src, unsigned short* __restrict__ dst, int n4) {
  int i = blockIdx.x * 256 + threadIdx.x;
  if (i < n4) {
    float4 f = ((const float4*)src)[i];
    ushort4 o;
    o.x = f2bf(f.x); o.y = f2bf(f.y); o.z = f2bf(f.z); o.w = f2bf(f.w);
    ((ushort4*)dst)[i] = o;
  }
}

// Zero both h double-buffers (fp32 state + bf16 WMMA operand copy). n = B*H.
__global__ __launch_bounds__(256) void init_state_kernel(
    float* __restrict__ h0f, float* __restrict__ h1f,
    unsigned short* __restrict__ h0b, unsigned short* __restrict__ h1b) {
  int i = blockIdx.x * 256 + threadIdx.x;
  h0f[i] = 0.0f; h1f[i] = 0.0f; h0b[i] = 0; h1b[i] = 0;
}

// ---------------------------------------------------------------------------
// Phase 1: gi GEMM. C[BT=32768, G=3072] = A[BT,K] * W_ih[G,K]^T + b_ih.
// Wave tile: 16(M) x 64(N), 4 accumulators, K-loop of 32 (16x16x32 bf16 WMMA).
// Output stored bf16 in [T, B, G] order so each step reads a contiguous slab.
__global__ __launch_bounds__(256) void gi_gemm_kernel(
    const unsigned short* __restrict__ A,    // [B*T, K] bf16 (m_enc)
    const unsigned short* __restrict__ W,    // [G, K]  bf16 (W_ih)
    const float* __restrict__ b_ih,          // [G]
    unsigned short* __restrict__ gi) {       // [T, B, G] bf16
  const int lane = threadIdx.x & 31;
  const int half = lane >> 4;
  const int lr   = lane & 15;
  const int w    = blockIdx.x * 8 + (threadIdx.x >> 5);
  const int mt   = w & 2047;     // 2048 M-tiles
  const int ns   = w >> 11;      // 48 N-strips of 64
  const int m0   = mt << 4;
  const int n0   = ns << 6;

  const unsigned short* pA = A + (size_t)(m0 + lr) * K_;
  const v8f vz = {0.f, 0.f, 0.f, 0.f, 0.f, 0.f, 0.f, 0.f};
  v8f acc[4] = {vz, vz, vz, vz};

  for (int kb = 0; kb < K_; kb += 32) {
    v16bf a = load_a_frag(pA, kb, half);
#pragma unroll
    for (int j = 0; j < 4; ++j) {
      const unsigned short* pB =
          W + (size_t)(n0 + j * 16 + lr) * K_ + kb + half * 16;
      v16bf b = *(const v16bf*)pB;
      acc[j] = __builtin_amdgcn_wmma_f32_16x16x32_bf16(
          false, a, false, b, (short)0, acc[j], false, false);
    }
  }

#pragma unroll
  for (int j = 0; j < 4; ++j) {
    const int n = n0 + j * 16 + lr;
    const float bias = b_ih[n];
#pragma unroll
    for (int r = 0; r < 8; ++r) {
      const int m = m0 + r + half * 8;          // m = b*T + t
      const int b = m >> 9;
      const int t = m & (T_ - 1);
      gi[((size_t)t * B_ + b) * G_ + n] = f2bf(acc[j][r] + bias);
    }
  }
}

// ---------------------------------------------------------------------------
// Phase 2: one recurrent step. 256 waves: 4 batch-tiles x 64 column-tiles.
// Each wave computes the r/z/n 16x16 gh tiles for its (batch-tile, col-tile),
// then gates + h update + output store, all lane-local in the D layout.
__global__ __launch_bounds__(128) void gru_step_kernel(
    const unsigned short* __restrict__ Whh,      // [G, K] bf16
    const float* __restrict__ b_hh,              // [G]
    const unsigned short* __restrict__ gi,       // [T, B, G] bf16
    const float* __restrict__ h_src,             // [B, H] fp32 (read)
    const unsigned short* __restrict__ h_src_bf, // [B, H] bf16 (read, A-matrix)
    float* __restrict__ h_dst,                   // [B, H] fp32 (write)
    unsigned short* __restrict__ h_dst_bf,       // [B, H] bf16 (write)
    float* __restrict__ out,                     // [B, T, H] fp32
    int t) {
  const int lane = threadIdx.x & 31;
  const int half = lane >> 4;
  const int lr   = lane & 15;
  const int w    = blockIdx.x * 4 + (threadIdx.x >> 5);
  const int mb   = w & 3;        // 4 batch tiles of 16
  const int jt   = w >> 2;       // 64 column tiles of 16
  const int m0   = mb << 4;
  const int j0   = jt << 4;

  const unsigned short* pA = h_src_bf + (size_t)(m0 + lr) * H_;
  const v8f vz = {0.f, 0.f, 0.f, 0.f, 0.f, 0.f, 0.f, 0.f};
  v8f cr = vz, cz = vz, cn = vz;

  const unsigned short* pBr = Whh + (size_t)(0 * H_ + j0 + lr) * K_ + half * 16;
  const unsigned short* pBz = Whh + (size_t)(1 * H_ + j0 + lr) * K_ + half * 16;
  const unsigned short* pBn = Whh + (size_t)(2 * H_ + j0 + lr) * K_ + half * 16;

  for (int kb = 0; kb < K_; kb += 32) {
    v16bf a = load_a_frag(pA, kb, half);
    v16bf br = *(const v16bf*)(pBr + kb);
    v16bf bz = *(const v16bf*)(pBz + kb);
    v16bf bn = *(const v16bf*)(pBn + kb);
    cr = __builtin_amdgcn_wmma_f32_16x16x32_bf16(false, a, false, br, (short)0, cr, false, false);
    cz = __builtin_amdgcn_wmma_f32_16x16x32_bf16(false, a, false, bz, (short)0, cz, false, false);
    cn = __builtin_amdgcn_wmma_f32_16x16x32_bf16(false, a, false, bn, (short)0, cn, false, false);
  }

  const int col = j0 + lr;
  const float bhr = b_hh[col];
  const float bhz = b_hh[H_ + col];
  const float bhn = b_hh[2 * H_ + col];
  const unsigned short* gi_t = gi + (size_t)t * B_ * G_;

#pragma unroll
  for (int r = 0; r < 8; ++r) {
    const int b = m0 + r + half * 8;
    const size_t gbase = (size_t)b * G_ + col;
    const float ir = bf2f(gi_t[gbase]);
    const float iz = bf2f(gi_t[gbase + H_]);
    const float in = bf2f(gi_t[gbase + 2 * H_]);
    const float hp = h_src[(size_t)b * H_ + col];

    const float rg = sig_(ir + cr[r] + bhr);
    const float zg = sig_(iz + cz[r] + bhz);
    const float ng = tanh_(in + rg * (cn[r] + bhn));
    const float hn = (1.0f - zg) * ng + zg * hp;

    h_dst[(size_t)b * H_ + col]    = hn;
    h_dst_bf[(size_t)b * H_ + col] = f2bf(hn);
    out[((size_t)b * T_ + t) * H_ + col] = hn;
  }
}

// ---------------------------------------------------------------------------
extern "C" void kernel_launch(void* const* d_in, const int* in_sizes, int n_in,
                              void* d_out, int out_size, void* d_ws, size_t ws_size,
                              hipStream_t stream) {
  (void)in_sizes; (void)n_in; (void)out_size; (void)ws_size;
  const float* m_enc = (const float*)d_in[0];   // [B,T,H]
  const float* W_ih  = (const float*)d_in[1];   // [3H,H]
  const float* W_hh  = (const float*)d_in[2];   // [3H,H]
  const float* b_ih  = (const float*)d_in[3];   // [3H]
  const float* b_hh  = (const float*)d_in[4];   // [3H]
  float* out = (float*)d_out;

  // Workspace layout (bytes): gi bf16 [T,B,G] | Wih bf16 | Whh bf16 |
  //   m_enc bf16 | h fp32 x2 | h bf16 x2.  Total ~268 MiB.
  char* ws = (char*)d_ws;
  size_t off = 0;
  unsigned short* gi   = (unsigned short*)(ws + off); off += (size_t)T_ * B_ * G_ * 2;
  unsigned short* Wihb = (unsigned short*)(ws + off); off += (size_t)G_ * K_ * 2;
  unsigned short* Whhb = (unsigned short*)(ws + off); off += (size_t)G_ * K_ * 2;
  unsigned short* Ab   = (unsigned short*)(ws + off); off += (size_t)B_ * T_ * H_ * 2;
  float* h0f = (float*)(ws + off); off += (size_t)B_ * H_ * 4;
  float* h1f = (float*)(ws + off); off += (size_t)B_ * H_ * 4;
  unsigned short* h0b = (unsigned short*)(ws + off); off += (size_t)B_ * H_ * 2;
  unsigned short* h1b = (unsigned short*)(ws + off); off += (size_t)B_ * H_ * 2;

  // bf16 staging of A and weights.
  {
    int n4 = (B_ * T_ * H_) / 4;
    cvt_f32_bf16_kernel<<<n4 / 256, 256, 0, stream>>>(m_enc, Ab, n4);
  }
  {
    int n4 = (G_ * K_) / 4;
    cvt_f32_bf16_kernel<<<n4 / 256, 256, 0, stream>>>(W_ih, Wihb, n4);
    cvt_f32_bf16_kernel<<<n4 / 256, 256, 0, stream>>>(W_hh, Whhb, n4);
  }
  init_state_kernel<<<(B_ * H_) / 256, 256, 0, stream>>>(h0f, h1f, h0b, h1b);

  // Phase 1: 2048 M-tiles x 48 N-strips = 98304 waves, 8 waves/block.
  gi_gemm_kernel<<<12288, 256, 0, stream>>>(Ab, Wihb, b_ih, gi);

  // Phase 2: sequential scan; stream order carries the t -> t+1 dependency.
  for (int t = 0; t < T_; ++t) {
    const float* hs          = (t & 1) ? h1f : h0f;
    float* hd                = (t & 1) ? h0f : h1f;
    const unsigned short* hsb = (t & 1) ? h1b : h0b;
    unsigned short* hdb      = (t & 1) ? h0b : h1b;
    gru_step_kernel<<<64, 128, 0, stream>>>(Whhb, b_hh, gi, hs, hsb, hd, hdb, out, t);
  }
}
